// DPGRNN_CFC_78443282694881
// MI455X (gfx1250) — compile-verified, matrix operand order checked by reading, not run on hardware
//
#include <hip/hip_runtime.h>
#include <hip/hip_bf16.h>

// ---------------------------------------------------------------------------
// DPGRNN / CfC for MI455X (gfx1250, wave32, WMMA f32_16x16x32_f16)
//
// Scans are latency-bound (161+250 serial steps); each step is a small batched
// GEMM.  Weight matrices stay resident in VGPRs as pre-gathered WMMA
// B-fragments for the entire scan; the 16-sequence hidden-state tile lives in
// LDS (f16 — it is only ever consumed as a WMMA A input); the CfC
// nonlinearity runs branch-free on the f32 WMMA accumulators using
// v_exp_f32 / v_rcp_f32 so it co-issues with the matrix pipe.
// ---------------------------------------------------------------------------

typedef __attribute__((ext_vector_type(16))) _Float16 v16h;
typedef __attribute__((ext_vector_type(8)))  float    v8f;

namespace dp {
constexpr int kB  = 8;
constexpr int kC  = 64;
constexpr int kT  = 250;
constexpr int kF  = 161;
constexpr int kH  = 64;
constexpr int kHh = 32;
constexpr int kBB = 128;
constexpr int kN1 = kB * kT;   // 2000 intra sequences
constexpr int kN2 = kB * kF;   // 1288 inter sequences
}

// ---- WMMA fragment helpers (layouts per CDNA5 ISA 7.12.2) ------------------

// 16-bit A matrix 16x32: lane L holds row M=L%16; VGPR v holds packed K pair;
// lanes 16..31 carry K offset +8; VGPRs 4..7 carry K offset +16.
__device__ __forceinline__ v16h frag_a_f16(const _Float16* __restrict__ p,
                                           int ld, int lane) {
  const int grp = (lane >> 4) & 1;
  const int m   = lane & 15;
  v16h a;
#pragma unroll
  for (int e = 0; e < 16; ++e) {
    const int v   = e >> 1;
    const int sub = e & 1;
    const int k   = ((v & 4) << 2) + ((v & 3) * 2 + sub) + grp * 8;
    a[e] = p[m * ld + k];
  }
  return a;
}

// B matrix 32x16 (K x N, row-major fp32 source, converted to f16 on load).
__device__ __forceinline__ v16h frag_b_f32(const float* __restrict__ p,
                                           int ld, int lane) {
  const int grp = (lane >> 4) & 1;
  const int n   = lane & 15;
  v16h b;
#pragma unroll
  for (int e = 0; e < 16; ++e) {
    const int v   = e >> 1;
    const int sub = e & 1;
    const int k   = ((v & 4) << 2) + ((v & 3) * 2 + sub) + grp * 8;
    b[e] = (_Float16)p[k * ld + n];
  }
  return b;
}

__device__ __forceinline__ v8f wmma16(v16h a, v16h b, v8f c) {
  return __builtin_amdgcn_wmma_f32_16x16x32_f16(
      /*neg_a=*/false, a, /*neg_b=*/false, b,
      /*c_mod=*/(short)0, c, /*reuse_a=*/false, /*reuse_b=*/false);
}

// lecun_tanh(x) = 1.7159*tanh(0.666x) = 1.7159 - 3.4318/(e^{1.332x}+1).
// Branch-free: v_exp_f32 saturates (+inf / 0) giving the correct +-1.7159
// limits, v_rcp_f32 avoids the precise-division EXEC-branch sequence.
__device__ __forceinline__ float lecun_tanh(float x) {
  const float e = __expf(1.332f * x);
  return 1.7159f - 3.4318f * __builtin_amdgcn_rcpf(e + 1.0f);
}

// ---- kernel 1: x (B,C,T,F) -> xp (B*T, F, C) in f32 + f16 ------------------
__global__ void k_transpose(const float* __restrict__ x,
                            float* __restrict__ xp,
                            _Float16* __restrict__ xp16) {
  using namespace dp;
  const size_t total = (size_t)kB * kT * kF * kC;
  const size_t idx = (size_t)blockIdx.x * blockDim.x + threadIdx.x;
  if (idx >= total) return;
  const int c = (int)(idx & 63);
  size_t r = idx >> 6;
  const int f = (int)(r % kF); r /= kF;
  const int t = (int)(r % kT); r /= kT;
  const int b = (int)r;
  const float v = x[(((size_t)b * kC + c) * kT + t) * kF + f];
  xp[idx]   = v;
  xp16[idx] = (_Float16)v;
}

// ---- kernel 2: intra bidirectional CfC scan over frequency -----------------
// grid = 125 tiles * 2 dirs; block = 128 threads (4 waves).
// Per step: z = [x_t(16x64) | h(16x32)] (LDS f16), backbone 16x128 via WMMA
// (wave w owns column tiles 2w,2w+1; Wb fragments register-resident), then
// ff1 16x32 via WMMA on waves 0,1, then CfC state update.
__global__ __launch_bounds__(128) void k_intra_scan(
    const _Float16* __restrict__ xp16,
    const float* __restrict__ Wb_f, const float* __restrict__ bb_f,
    const float* __restrict__ W1_f, const float* __restrict__ b1_f,
    const float* __restrict__ wt_f, const float* __restrict__ A_f,
    const float* __restrict__ Wb_b, const float* __restrict__ bb_b,
    const float* __restrict__ W1_b, const float* __restrict__ b1_b,
    const float* __restrict__ wt_b, const float* __restrict__ A_b,
    _Float16* __restrict__ ycat) {
  using namespace dp;
  constexpr int ZS = 104;  // 96 + pad
  constexpr int GS = 136;  // 128 + pad
  __shared__ _Float16 zbuf[16 * ZS];
  __shared__ _Float16 gbuf[16 * GS];

  const int tile = blockIdx.x >> 1;
  const int dir  = blockIdx.x & 1;
  const int seq0 = tile * 16;
  const int tid  = threadIdx.x;
  const int w    = tid >> 5;
  const int lane = tid & 31;
  const int grp  = lane >> 4;
  const int ln   = lane & 15;

  const float* Wb = dir ? Wb_b : Wb_f;
  const float* bb = dir ? bb_b : bb_f;
  const float* W1 = dir ? W1_b : W1_f;
  const float* b1 = dir ? b1_b : b1_f;
  const float* wt = dir ? wt_b : wt_f;
  const float* Am = dir ? A_b  : A_f;

  // Weight fragments live in VGPRs for the whole scan.
  v16h WbF[2][3];
#pragma unroll
  for (int c = 0; c < 2; ++c)
#pragma unroll
    for (int kt = 0; kt < 3; ++kt)
      WbF[c][kt] =
          frag_b_f32(Wb + (size_t)kt * 32 * kBB + (2 * w + c) * 16, kBB, lane);

  float bbv[2];
  bbv[0] = bb[(2 * w + 0) * 16 + ln];
  bbv[1] = bb[(2 * w + 1) * 16 + ln];

  v16h W1F[4] = {};
  float b1v = 0.f, Aval = 0.f, wtv = 0.f;
  if (w < 2) {
#pragma unroll
    for (int kt = 0; kt < 4; ++kt)
      W1F[kt] = frag_b_f32(W1 + (size_t)kt * 32 * kHh + w * 16, kHh, lane);
    const int n = w * 16 + ln;
    b1v  = b1[n];
    Aval = Am[n];
    wtv  = fabsf(wt[n]);
  }

  for (int i = tid; i < 16 * ZS; i += 128) zbuf[i] = (_Float16)0.0f;  // h0 = 0

  for (int s = 0; s < kF; ++s) {
    const int f = dir ? (kF - 1 - s) : s;
    __syncthreads();  // prior h writes done; safe to restage x
    for (int i = tid; i < 16 * kC; i += 128) {
      const int r = i >> 6, c = i & 63;
      zbuf[r * ZS + c] = xp16[((size_t)(seq0 + r) * kF + f) * kC + c];
    }
    if (s + 1 < kF) {
      const int fn = dir ? (kF - 2 - s) : (s + 1);
      __builtin_prefetch(
          &xp16[((size_t)(seq0 + (tid >> 3)) * kF + fn) * kC + (tid & 7) * 8],
          0, 1);
    }
    __syncthreads();

    v16h aF[3];
#pragma unroll
    for (int kt = 0; kt < 3; ++kt) aF[kt] = frag_a_f16(zbuf + kt * 32, ZS, lane);

#pragma unroll
    for (int c = 0; c < 2; ++c) {
      v8f acc;
#pragma unroll
      for (int r = 0; r < 8; ++r) acc[r] = bbv[c];
#pragma unroll
      for (int kt = 0; kt < 3; ++kt) acc = wmma16(aF[kt], WbF[c][kt], acc);
#pragma unroll
      for (int r = 0; r < 8; ++r) {
        const float gv = lecun_tanh(acc[r]);
        gbuf[(r + 8 * grp) * GS + (2 * w + c) * 16 + ln] = (_Float16)gv;
      }
    }
    __syncthreads();

    if (w < 2) {  // ff1: 16x32, waves 0/1 each own a 16-col tile
      v8f acc;
#pragma unroll
      for (int r = 0; r < 8; ++r) acc[r] = b1v;
#pragma unroll
      for (int kt = 0; kt < 4; ++kt) {
        const v16h ga = frag_a_f16(gbuf + kt * 32, GS, lane);
        acc = wmma16(ga, W1F[kt], acc);
      }
#pragma unroll
      for (int r = 0; r < 8; ++r) {
        const float ff1 = acc[r];
        const float hn  = -Aval * __expf(-(wtv + fabsf(ff1))) * ff1 + Aval;
        const int m = r + 8 * grp;
        zbuf[m * ZS + kC + w * 16 + ln] = (_Float16)hn;  // recurrent state
        ycat[((size_t)(seq0 + m) * kF + s) * kH + dir * kHh + w * 16 + ln] =
            (_Float16)hn;  // note: reverse branch NOT flipped back (faithful)
      }
    }
  }
}

// ---- kernel 3: intra FC, y(322000x64) @ W(64x64) + b -----------------------
__global__ __launch_bounds__(128) void k_fc_intra(
    const _Float16* __restrict__ Ain, const float* __restrict__ Wf,
    const float* __restrict__ bf, float* __restrict__ out) {
  using namespace dp;
  const int row0 = blockIdx.x * 16;
  const int tid = threadIdx.x, w = tid >> 5, lane = tid & 31;
  const int grp = lane >> 4, ln = lane & 15;
  v16h WF[2];
#pragma unroll
  for (int kt = 0; kt < 2; ++kt)
    WF[kt] = frag_b_f32(Wf + (size_t)kt * 32 * kH + w * 16, kH, lane);
  v16h aF[2];
#pragma unroll
  for (int kt = 0; kt < 2; ++kt)
    aF[kt] = frag_a_f16(Ain + (size_t)row0 * kH + kt * 32, kH, lane);
  v8f acc;
  const float bv = bf[w * 16 + ln];
#pragma unroll
  for (int r = 0; r < 8; ++r) acc[r] = bv;
#pragma unroll
  for (int kt = 0; kt < 2; ++kt) acc = wmma16(aF[kt], WF[kt], acc);
#pragma unroll
  for (int r = 0; r < 8; ++r)
    out[(size_t)(row0 + r + 8 * grp) * kH + w * 16 + ln] = acc[r];
}

// ---- kernel 4: intra LN over (F,H) + residual; repack f16 (B*F,T,H) --------
__global__ __launch_bounds__(256) void k_intra_ln(
    const float* __restrict__ fc, const float* __restrict__ xp,
    const float* __restrict__ gam, const float* __restrict__ bet,
    float* __restrict__ intra_out, _Float16* __restrict__ xin16) {
  using namespace dp;
  __shared__ float r1[256], r2[256];
  const int bt = blockIdx.x;
  const int b = bt / kT, t = bt % kT;
  const size_t base = (size_t)bt * kF * kH;
  float s = 0.f, s2 = 0.f;
  for (int i = threadIdx.x; i < kF * kH; i += 256) {
    const float v = fc[base + i];
    s += v; s2 += v * v;
  }
  r1[threadIdx.x] = s; r2[threadIdx.x] = s2;
  __syncthreads();
  for (int o = 128; o > 0; o >>= 1) {
    if (threadIdx.x < o) {
      r1[threadIdx.x] += r1[threadIdx.x + o];
      r2[threadIdx.x] += r2[threadIdx.x + o];
    }
    __syncthreads();
  }
  const float inv_n = 1.0f / (float)(kF * kH);
  const float mu  = r1[0] * inv_n;
  const float var = r2[0] * inv_n - mu * mu;
  const float inv = rsqrtf(var + 1e-8f);
  for (int i = threadIdx.x; i < kF * kH; i += 256) {
    const float nrm = (fc[base + i] - mu) * inv * gam[i] + bet[i];
    const float o = xp[base + i] + nrm;
    intra_out[base + i] = o;
    const int f = i >> 6, h = i & 63;
    xin16[(((size_t)(b * kF + f)) * kT + t) * kH + h] = (_Float16)o;
  }
}

// ---- kernel 5: inter CfC scan over time ------------------------------------
// grid = ceil(1288/16)=81 tiles; z = [x_t(16x64) | h(16x64)] -> 16x128.
__global__ __launch_bounds__(128) void k_inter_scan(
    const _Float16* __restrict__ xin,
    const float* __restrict__ Wb, const float* __restrict__ bb,
    const float* __restrict__ W1, const float* __restrict__ b1,
    const float* __restrict__ wt, const float* __restrict__ Am,
    _Float16* __restrict__ yout) {
  using namespace dp;
  constexpr int ZS = 136;  // 128 + pad
  constexpr int GS = 136;
  __shared__ _Float16 zbuf[16 * ZS];
  __shared__ _Float16 gbuf[16 * GS];

  const int seq0 = blockIdx.x * 16;
  const int tid = threadIdx.x, w = tid >> 5, lane = tid & 31;
  const int grp = lane >> 4, ln = lane & 15;

  v16h WbF[2][4];
#pragma unroll
  for (int c = 0; c < 2; ++c)
#pragma unroll
    for (int kt = 0; kt < 4; ++kt)
      WbF[c][kt] =
          frag_b_f32(Wb + (size_t)kt * 32 * kBB + (2 * w + c) * 16, kBB, lane);
  v16h W1F[4];
#pragma unroll
  for (int kt = 0; kt < 4; ++kt)
    W1F[kt] = frag_b_f32(W1 + (size_t)kt * 32 * kH + w * 16, kH, lane);

  float bbv[2];
  bbv[0] = bb[(2 * w + 0) * 16 + ln];
  bbv[1] = bb[(2 * w + 1) * 16 + ln];
  const int nh = w * 16 + ln;
  const float b1v  = b1[nh];
  const float Aval = Am[nh];
  const float wtv  = fabsf(wt[nh]);

  for (int i = tid; i < 16 * ZS; i += 128) zbuf[i] = (_Float16)0.0f;

  for (int t = 0; t < kT; ++t) {
    __syncthreads();
    for (int i = tid; i < 16 * kH; i += 128) {
      const int r = i >> 6, c = i & 63;
      const int seq = seq0 + r;
      zbuf[r * ZS + c] = (seq < kN2)
          ? xin[((size_t)seq * kT + t) * kH + c] : (_Float16)0.0f;
    }
    if (t + 1 < kT) {
      const int seqp = seq0 + (tid >> 3);
      if (seqp < kN2)
        __builtin_prefetch(
            &xin[((size_t)seqp * kT + t + 1) * kH + (tid & 7) * 8], 0, 1);
    }
    __syncthreads();

    v16h aF[4];
#pragma unroll
    for (int kt = 0; kt < 4; ++kt) aF[kt] = frag_a_f16(zbuf + kt * 32, ZS, lane);

#pragma unroll
    for (int c = 0; c < 2; ++c) {
      v8f acc;
#pragma unroll
      for (int r = 0; r < 8; ++r) acc[r] = bbv[c];
#pragma unroll
      for (int kt = 0; kt < 4; ++kt) acc = wmma16(aF[kt], WbF[c][kt], acc);
#pragma unroll
      for (int r = 0; r < 8; ++r) {
        const float gv = lecun_tanh(acc[r]);
        gbuf[(r + 8 * grp) * GS + (2 * w + c) * 16 + ln] = (_Float16)gv;
      }
    }
    __syncthreads();

    {  // ff1: 16x64, each wave owns one 16-col tile
      v8f acc;
#pragma unroll
      for (int r = 0; r < 8; ++r) acc[r] = b1v;
#pragma unroll
      for (int kt = 0; kt < 4; ++kt) {
        const v16h ga = frag_a_f16(gbuf + kt * 32, GS, lane);
        acc = wmma16(ga, W1F[kt], acc);
      }
#pragma unroll
      for (int r = 0; r < 8; ++r) {
        const float ff1 = acc[r];
        const float hn  = -Aval * __expf(-(wtv + fabsf(ff1))) * ff1 + Aval;
        const int m = r + 8 * grp;
        zbuf[m * ZS + kH + w * 16 + ln] = (_Float16)hn;
        const int seq = seq0 + m;
        if (seq < kN2)
          yout[((size_t)seq * kT + t) * kH + w * 16 + ln] = (_Float16)hn;
      }
    }
  }
}

// ---- kernel 6: inter FC with transposed store (B,F,T,H)->(B,T,F,H) ---------
__global__ __launch_bounds__(128) void k_fc_inter(
    const _Float16* __restrict__ Ain, const float* __restrict__ Wf,
    const float* __restrict__ bf, float* __restrict__ out) {
  using namespace dp;
  const int row0 = blockIdx.x * 16;
  const int tid = threadIdx.x, w = tid >> 5, lane = tid & 31;
  const int grp = lane >> 4, ln = lane & 15;
  v16h WF[2];
#pragma unroll
  for (int kt = 0; kt < 2; ++kt)
    WF[kt] = frag_b_f32(Wf + (size_t)kt * 32 * kH + w * 16, kH, lane);
  v16h aF[2];
#pragma unroll
  for (int kt = 0; kt < 2; ++kt)
    aF[kt] = frag_a_f16(Ain + (size_t)row0 * kH + kt * 32, kH, lane);
  v8f acc;
  const float bv = bf[w * 16 + ln];
#pragma unroll
  for (int r = 0; r < 8; ++r) acc[r] = bv;
#pragma unroll
  for (int kt = 0; kt < 2; ++kt) acc = wmma16(aF[kt], WF[kt], acc);
#pragma unroll
  for (int r = 0; r < 8; ++r) {
    const int row = row0 + r + 8 * grp;  // row = (b*F + f)*T + t
    const int b   = row / (kF * kT);
    const int rem = row % (kF * kT);
    const int f   = rem / kT;
    const int t   = rem % kT;
    out[(((size_t)b * kT + t) * kF + f) * kH + w * 16 + ln] = acc[r];
  }
}

// ---- kernel 7: inter LN + residual + final (B,H,T,F) transpose -------------
__global__ __launch_bounds__(256) void k_inter_ln(
    const float* __restrict__ fc, const float* __restrict__ intra_out,
    const float* __restrict__ gam, const float* __restrict__ bet,
    float* __restrict__ out) {
  using namespace dp;
  __shared__ float r1[256], r2[256];
  const int bt = blockIdx.x;
  const int b = bt / kT, t = bt % kT;
  const size_t base = (size_t)bt * kF * kH;
  float s = 0.f, s2 = 0.f;
  for (int i = threadIdx.x; i < kF * kH; i += 256) {
    const float v = fc[base + i];
    s += v; s2 += v * v;
  }
  r1[threadIdx.x] = s; r2[threadIdx.x] = s2;
  __syncthreads();
  for (int o = 128; o > 0; o >>= 1) {
    if (threadIdx.x < o) {
      r1[threadIdx.x] += r1[threadIdx.x + o];
      r2[threadIdx.x] += r2[threadIdx.x + o];
    }
    __syncthreads();
  }
  const float inv_n = 1.0f / (float)(kF * kH);
  const float mu  = r1[0] * inv_n;
  const float var = r2[0] * inv_n - mu * mu;
  const float inv = rsqrtf(var + 1e-8f);
  for (int i = threadIdx.x; i < kF * kH; i += 256) {
    const float nrm = (fc[base + i] - mu) * inv * gam[i] + bet[i];
    const float o = intra_out[base + i] + nrm;
    const int f = i >> 6, h = i & 63;
    out[(((size_t)b * kH + h) * kT + t) * kF + f] = o;
  }
}

// ---------------------------------------------------------------------------
extern "C" void kernel_launch(void* const* d_in, const int* in_sizes, int n_in,
                              void* d_out, int out_size, void* d_ws,
                              size_t ws_size, hipStream_t stream) {
  using namespace dp;
  (void)in_sizes; (void)n_in; (void)out_size; (void)ws_size;

  const float* x       = (const float*)d_in[0];
  const float* i1_Wb   = (const float*)d_in[1];
  const float* i1_bb   = (const float*)d_in[2];
  const float* i1_W1   = (const float*)d_in[3];
  const float* i1_b1   = (const float*)d_in[4];
  const float* i1_wtau = (const float*)d_in[5];
  const float* i1_A    = (const float*)d_in[6];
  const float* i2_Wb   = (const float*)d_in[7];
  const float* i2_bb   = (const float*)d_in[8];
  const float* i2_W1   = (const float*)d_in[9];
  const float* i2_b1   = (const float*)d_in[10];
  const float* i2_wtau = (const float*)d_in[11];
  const float* i2_A    = (const float*)d_in[12];
  const float* r_Wb    = (const float*)d_in[13];
  const float* r_bb    = (const float*)d_in[14];
  const float* r_W1    = (const float*)d_in[15];
  const float* r_b1    = (const float*)d_in[16];
  const float* r_wtau  = (const float*)d_in[17];
  const float* r_A     = (const float*)d_in[18];
  const float* intra_fc_W = (const float*)d_in[19];
  const float* intra_fc_b = (const float*)d_in[20];
  const float* intra_ln_g = (const float*)d_in[21];
  const float* intra_ln_b = (const float*)d_in[22];
  const float* inter_fc_W = (const float*)d_in[23];
  const float* inter_fc_b = (const float*)d_in[24];
  const float* inter_ln_g = (const float*)d_in[25];
  const float* inter_ln_b = (const float*)d_in[26];

  char* ws = (char*)d_ws;
  size_t off = 0;
  auto take = [&](size_t bytes) -> char* {
    char* p = ws + off;
    off += (bytes + 255) & ~(size_t)255;
    return p;
  };

  const size_t NE = (size_t)kB * kT * kF * kC;  // 20,608,000 (C==H==64)
  float*    xp        = (float*)take(NE * sizeof(float));
  _Float16* xp16      = (_Float16*)take(NE * sizeof(_Float16));
  _Float16* ycat      = (_Float16*)take(NE * sizeof(_Float16));
  float*    fc1       = (float*)take(NE * sizeof(float));
  float*    intra_out = (float*)take(NE * sizeof(float));
  _Float16* xin16     = (_Float16*)take((size_t)kN2 * kT * kH * sizeof(_Float16));
  _Float16* ivy       = (_Float16*)take((size_t)kN2 * kT * kH * sizeof(_Float16));
  float*    fc2       = (float*)take(NE * sizeof(float));

  const int rows = kN1 * kF;  // 322000 == kN2 * kT
  const int fc_blocks = rows / 16;

  k_transpose<<<(int)((NE + 255) / 256), 256, 0, stream>>>(x, xp, xp16);
  k_intra_scan<<<(kN1 / 16) * 2, 128, 0, stream>>>(
      xp16, i1_Wb, i1_bb, i1_W1, i1_b1, i1_wtau, i1_A,
      i2_Wb, i2_bb, i2_W1, i2_b1, i2_wtau, i2_A, ycat);
  k_fc_intra<<<fc_blocks, 128, 0, stream>>>(ycat, intra_fc_W, intra_fc_b, fc1);
  k_intra_ln<<<kB * kT, 256, 0, stream>>>(fc1, xp, intra_ln_g, intra_ln_b,
                                          intra_out, xin16);
  k_inter_scan<<<(kN2 + 15) / 16, 128, 0, stream>>>(
      xin16, r_Wb, r_bb, r_W1, r_b1, r_wtau, r_A, ivy);
  k_fc_inter<<<fc_blocks, 128, 0, stream>>>(ivy, inter_fc_W, inter_fc_b, fc2);
  k_inter_ln<<<kB * kT, 256, 0, stream>>>(fc2, intra_out, inter_ln_g,
                                          inter_ln_b, (float*)d_out);
}
